// HierarchicalLayer_15607911153870
// MI455X (gfx1250) — matrix operand order repaired and checked by reading, not compiled, not run on hardware
//
#include <hip/hip_runtime.h>

typedef __attribute__((ext_vector_type(16))) _Float16 v16h;
typedef __attribute__((ext_vector_type(8)))  _Float16 v8h;
typedef __attribute__((ext_vector_type(4)))  _Float16 v4h;
typedef __attribute__((ext_vector_type(8)))  float    v8f;
typedef __attribute__((ext_vector_type(4)))  float    v4f;

#define DHID    128
#define PITCH   136          // f16 LDS pitch; 272B rows -> 16B aligned, conflict-free
#define EPB     64           // edges per block
#define THREADS 256

// ---------------- zero-fill d_out (harness poisons it; we accumulate) ------
__global__ void zero_out_kernel(float* __restrict__ out, int n4) {
    int i = blockIdx.x * blockDim.x + threadIdx.x;
    if (i < n4) ((v4f*)out)[i] = (v4f){0.f, 0.f, 0.f, 0.f};
}

// ---------------- one-shot: W (fp32 row-major [k][n]) -> f16 W^T [n][k] ----
__global__ void prep_weights_kernel(const float* __restrict__ W1,
                                    const float* __restrict__ W2,
                                    _Float16* __restrict__ w1t,
                                    _Float16* __restrict__ w2t) {
    int i = blockIdx.x * blockDim.x + threadIdx.x;   // 0 .. 16383
    if (i < DHID * DHID) {
        int k = i >> 7;
        int n = i & (DHID - 1);
        w1t[n * DHID + k] = (_Float16)W1[i];
        w2t[n * DHID + k] = (_Float16)W2[i];
    }
}

// fast SiLU: x * rcp(1 + exp(-x))  (v_exp_f32 + v_rcp_f32, no IEEE div chain)
__device__ __forceinline__ float fast_silu(float x) {
    float t = __expf(-x);
    return x * __builtin_amdgcn_rcpf(1.0f + t);
}

// ---------------- fragment loaders ----------------------------------------
// A-matrix 16x32 f16 (ISA 7.12.2): lane L holds row M=L%16.
// lanes 0-15: K = {0..7, 16..23}; lanes 16-31: K = {8..15, 24..31}.
__device__ __forceinline__ v16h load_a_frag(const _Float16* __restrict__ mat,
                                            int row, int k0, int lhalf) {
    const _Float16* p = mat + row * PITCH + k0 + 8 * lhalf;
    v8h lo = *(const v8h*)(p);
    v8h hi = *(const v8h*)(p + 16);
    v16h a;
#pragma unroll
    for (int i = 0; i < 8; ++i) { a[i] = lo[i]; a[i + 8] = hi[i]; }
    return a;
}

// B-matrix 32x16 f16 (column n = lane%16): lanes 0-15 hold K={0..7,16..23},
// lanes 16-31 hold K={8..15,24..31}. Loaded straight from L2-resident global W^T.
__device__ __forceinline__ v16h load_b_frag_g(const _Float16* __restrict__ wt,
                                              int col, int k0, int lhalf) {
    const _Float16* p = wt + col * DHID + k0 + 8 * lhalf;
    v8h lo = *(const v8h*)(p);
    v8h hi = *(const v8h*)(p + 16);
    v16h b;
#pragma unroll
    for (int i = 0; i < 8; ++i) { b[i] = lo[i]; b[i + 8] = hi[i]; }
    return b;
}

// guaranteed native no-return f32 atomic (L2 near-atomic, STOREcnt-tracked)
__device__ __forceinline__ void atomic_add_f32_global(float* p, float v) {
    asm volatile("global_atomic_add_f32 %0, %1, off"
                 :
                 : "v"(p), "v"(v)
                 : "memory");
}

// ---------------- fused gather -> MLP (WMMA f16) -> mask -> scatter-add ----
__global__ __launch_bounds__(THREADS)
void edge_mlp_scatter_kernel(const long long* __restrict__ edge_index, // [2,E]
                             const float* __restrict__ node_emb,       // [N,D]
                             const float* __restrict__ group_emb,      // [N,D]
                             const float* __restrict__ edge_w,         // [E]
                             const float* __restrict__ scale_factor,   // [1]
                             const _Float16* __restrict__ w1t,         // [D,D] f16 W1^T
                             const float* __restrict__ b1,             // [D]
                             const _Float16* __restrict__ w2t,         // [D,D] f16 W2^T
                             const float* __restrict__ b2,             // [D]
                             float* __restrict__ out,                  // [N,D]
                             int E) {
    __shared__ _Float16 comb[EPB * PITCH];   // gathered inputs  [m][k]
    __shared__ _Float16 h1  [EPB * PITCH];   // SiLU(GEMM1)      [m][k]
    __shared__ int   srcI[EPB];
    __shared__ int   dstI[EPB];
    __shared__ float maskf[EPB];

    const int tid = threadIdx.x;
    const int e0  = blockIdx.x * EPB;

    // ---- edge metadata ----
    const float thr = 0.5f * scale_factor[0];
    for (int m = tid; m < EPB; m += THREADS) {
        int e = e0 + m;
        if (e < E) {
            srcI[m]  = (int)edge_index[e];
            dstI[m]  = (int)edge_index[(long long)E + e];
            maskf[m] = (edge_w[e] < thr) ? 1.0f : 0.0f;
        } else {
            srcI[m] = 0; dstI[m] = 0; maskf[m] = 0.0f;
        }
    }
    __syncthreads();

    // ---- gather: combined = node[src] + group[dst], float4 loads ----
    // EPB*DHID/4 = 2048 quads, 8 per thread; one row per 32 consecutive threads
    for (int q = tid; q < EPB * (DHID / 4); q += THREADS) {
        int m  = q >> 5;             // edge row
        int k4 = (q & 31) << 2;      // column (multiple of 4)
        v4f a = *(const v4f*)(node_emb  + (long long)srcI[m] * DHID + k4);
        v4f b = *(const v4f*)(group_emb + (long long)dstI[m] * DHID + k4);
        v4h h;
#pragma unroll
        for (int i = 0; i < 4; ++i) h[i] = (_Float16)(a[i] + b[i]);
        *(v4h*)(comb + m * PITCH + k4) = h;   // 8B ds_store
    }
    __syncthreads();

    const int lane  = tid & 31;
    const int wv    = tid >> 5;       // wave id -> N tile (8 waves x 16 cols)
    const int lhalf = lane >> 4;
    const int lmod  = lane & 15;
    const int n     = wv * 16 + lmod; // this lane's output column

    // ================= GEMM1: h1 = SiLU(comb @ W1 + b1) =================
    v16h bw[4];
#pragma unroll
    for (int ks = 0; ks < 4; ++ks) bw[ks] = load_b_frag_g(w1t, n, 32 * ks, lhalf);
    const float bias1 = b1[n];

#pragma unroll
    for (int mt = 0; mt < 4; ++mt) {
        v8f acc = {};
#pragma unroll
        for (int ks = 0; ks < 4; ++ks) {
            v16h a = load_a_frag(comb, mt * 16 + lmod, 32 * ks, lhalf);
            acc = __builtin_amdgcn_wmma_f32_16x16x32_f16(
                false, a, false, bw[ks], (short)0, acc, false, false);
        }
        // C/D layout: VGPR v, lanes 0-15 -> M=v, lanes 16-31 -> M=v+8
#pragma unroll
        for (int v = 0; v < 8; ++v) {
            float s = fast_silu(acc[v] + bias1);
            h1[(mt * 16 + v + 8 * lhalf) * PITCH + n] = (_Float16)s;
        }
    }
    __syncthreads();

    // ================= GEMM2: h = h1 @ W2 + b2, mask, scatter ===========
#pragma unroll
    for (int ks = 0; ks < 4; ++ks) bw[ks] = load_b_frag_g(w2t, n, 32 * ks, lhalf);
    const float bias2 = b2[n];

#pragma unroll
    for (int mt = 0; mt < 4; ++mt) {
        v8f acc = {};
#pragma unroll
        for (int ks = 0; ks < 4; ++ks) {
            v16h a = load_a_frag(h1, mt * 16 + lmod, 32 * ks, lhalf);
            acc = __builtin_amdgcn_wmma_f32_16x16x32_f16(
                false, a, false, bw[ks], (short)0, acc, false, false);
        }
#pragma unroll
        for (int v = 0; v < 8; ++v) {
            int   ml = mt * 16 + v + 8 * lhalf;     // local edge row
            float mv = maskf[ml];
            float val = (acc[v] + bias2) * mv;
            if (mv != 0.0f) {
                atomic_add_f32_global(out + (long long)srcI[ml] * DHID + n, val);
            }
        }
    }
}

// --------------------------------------------------------------------------
extern "C" void kernel_launch(void* const* d_in, const int* in_sizes, int n_in,
                              void* d_out, int out_size, void* d_ws, size_t ws_size,
                              hipStream_t stream) {
    const long long* edge_index  = (const long long*)d_in[0]; // int64 [2,E]
    const float*     node_emb    = (const float*)d_in[1];     // [N,128]
    const float*     group_emb   = (const float*)d_in[2];     // [N,128]
    const float*     edge_w      = (const float*)d_in[3];     // [E]
    // d_in[4] = edge_vec (unused by the reference computation)
    const float*     scale_fac   = (const float*)d_in[5];     // [1]
    const float*     W1          = (const float*)d_in[6];     // [128,128]
    const float*     b1          = (const float*)d_in[7];     // [128]
    const float*     W2          = (const float*)d_in[8];     // [128,128]
    const float*     b2          = (const float*)d_in[9];     // [128]
    float*           out         = (float*)d_out;             // [N,128]

    const int E = in_sizes[3];

    // f16 transposed weights live in workspace (64 KB), rebuilt every launch
    _Float16* w1t = (_Float16*)d_ws;
    _Float16* w2t = w1t + DHID * DHID;

    hipLaunchKernelGGL(prep_weights_kernel, dim3((DHID * DHID + 255) / 256),
                       dim3(256), 0, stream, W1, W2, w1t, w2t);

    // zero the accumulation target every launch (deterministic)
    int n4 = out_size / 4;
    hipLaunchKernelGGL(zero_out_kernel, dim3((n4 + 255) / 256), dim3(256), 0,
                       stream, out, n4);

    int blocks = (E + EPB - 1) / EPB;
    hipLaunchKernelGGL(edge_mlp_scatter_kernel, dim3(blocks), dim3(THREADS),
                       0, stream,
                       edge_index, node_emb, group_emb, edge_w, scale_fac,
                       w1t, b1, w2t, b2, out, E);
}
